// STBlock_34067680592025
// MI455X (gfx1250) — matrix operand order repaired and checked by reading, not compiled, not run on hardware
//
#include <hip/hip_runtime.h>
#include <hip/hip_bf16.h>
#include <math.h>
#include <stdint.h>

// ---------------------------------------------------------------------------
// Spatio-temporal transformer block for MI455X (gfx1250, wave32, WMMA).
// B=2, T=16, S=256, C=1024, H=16, hd=64, hidden=4096, NTOK = B*T*S = 8192.
// All big matmuls run on v_wmma_f32_16x16x32_bf16 (bf16 in, fp32 accumulate).
// GEMM: 64x128 block tile, double-buffered GLOBAL_LOAD_ASYNC_TO_LDS_B128
// (ASYNCcnt-pipelined DMA overlapped with WMMA). Attention: flash-style,
// K/V async-staged to LDS, online softmax, all matmuls on WMMA.
// ---------------------------------------------------------------------------

typedef __bf16 bf16;
typedef __attribute__((ext_vector_type(16))) __bf16 v16bf;
typedef __attribute__((ext_vector_type(8)))  float  v8f;

#define DEV __device__ __forceinline__

static constexpr int NTOK = 8192;   // B*T*S
static constexpr int C    = 1024;
static constexpr int H    = 16;
static constexpr int HD   = 64;
static constexpr int HID  = 4096;

// ---- bf16 <-> f32 helpers (pure bit ops) ----------------------------------
DEV unsigned short bf_bits(bf16 h) { union { bf16 h; unsigned short s; } u; u.h = h; return u.s; }
DEV bf16 bits_bf(unsigned short s) { union { bf16 h; unsigned short s; } u; u.s = s; return u.h; }
DEV float bf2f(bf16 h) { union { unsigned u; float f; } c; c.u = ((unsigned)bf_bits(h)) << 16; return c.f; }
DEV bf16 f2bf(float f) {
  union { float f; unsigned u; } c; c.f = f;
  unsigned u = c.u + 0x7FFFu + ((c.u >> 16) & 1u);   // round-to-nearest-even
  return bits_bf((unsigned short)(u >> 16));
}

// K-offset of pair i for A-side fragments (16x32 bf16, M x K), per ISA 7.12.2.
DEV int aK(int i, int half) { return ((i & 4) << 2) + (half << 3) + ((i & 3) << 1); }
// Contiguous K-offset of pair i (k = 16*half + 2i); unused, B fragments also use aK.
DEV int bK(int i, int half) { return (half << 4) + (i << 1); }

// ---------------------------------------------------------------------------
// Weight convert + transpose: W[K][N] f32  ->  WT[N][K] bf16
// ---------------------------------------------------------------------------
__global__ void wt_cvt(const float* __restrict__ W, bf16* __restrict__ WT, int K, int N) {
  const int n = blockIdx.x;
  for (int k = threadIdx.x; k < K; k += blockDim.x)
    WT[(size_t)n * K + k] = f2bf(W[(size_t)k * N + n]);
}

// ---------------------------------------------------------------------------
// Row LayerNorm over C=1024, one wave32 per token, output bf16.
// ---------------------------------------------------------------------------
__global__ void ln_rows(const float* __restrict__ X, const float* __restrict__ w,
                        const float* __restrict__ b, bf16* __restrict__ Y, int ntok) {
  const int lane = threadIdx.x & 31;
  const int tok  = blockIdx.x * (blockDim.x >> 5) + (threadIdx.x >> 5);
  if (tok >= ntok) return;
  const float* x = X + (size_t)tok * C;
  float v[32], s = 0.f, sq = 0.f;
#pragma unroll
  for (int i = 0; i < 32; ++i) { float t = x[lane + (i << 5)]; v[i] = t; s += t; sq += t * t; }
#pragma unroll
  for (int off = 16; off; off >>= 1) { s += __shfl_xor(s, off, 32); sq += __shfl_xor(sq, off, 32); }
  const float mean = s * (1.f / C);
  const float var  = sq * (1.f / C) - mean * mean;
  const float rs   = rsqrtf(var + 1e-5f);
  bf16* y = Y + (size_t)tok * C;
#pragma unroll
  for (int i = 0; i < 32; ++i) {
    const int ci = lane + (i << 5);
    y[ci] = f2bf((v[i] - mean) * rs * w[ci] + b[ci]);
  }
}

// ---------------------------------------------------------------------------
// LDS-blocked WMMA GEMM with double-buffered async DMA.
// out[M][N] = A[M][K](bf16) @ WT[N][K]^T(bf16), fp32 accumulate.
// Block: 256 threads = 8 waves as 2(M) x 4(N); block tile 64x128, K-step 32.
// Per K-slab each wave issues exactly 3 GLOBAL_LOAD_ASYNC_TO_LDS_B128 (1 A +
// 2 B). Next slab is issued before consuming the current one; async loads
// complete in order, so s_wait_asynccnt 3 guarantees the current slab landed.
// EPI 0: store bf16. EPI 1: +bias, exact GELU -> bf16. EPI 2: +bias +resid -> f32.
// ---------------------------------------------------------------------------
template <int EPI>
__global__ __launch_bounds__(256, 1)
void gemm_bf16(const bf16* __restrict__ A, const bf16* __restrict__ WT,
               const float* __restrict__ bias, const float* __restrict__ resid,
               void* __restrict__ outp, int M, int N, int K) {
  __shared__ __align__(16) bf16 As[2][64 * 32];
  __shared__ __align__(16) bf16 Bs[2][128 * 32];
  const int tid  = threadIdx.x;
  const int lane = tid & 31, wave = tid >> 5;
  const int half = lane >> 4, r16 = lane & 15;
  const int wm = wave >> 2, wn = wave & 3;           // wave grid 2 x 4
  const int m0 = blockIdx.y << 6, n0 = blockIdx.x << 7;

  const unsigned aBase[2] = { (unsigned)(uintptr_t)&As[0][0], (unsigned)(uintptr_t)&As[1][0] };
  const unsigned bBase[2] = { (unsigned)(uintptr_t)&Bs[0][0], (unsigned)(uintptr_t)&Bs[1][0] };

  // Issue one K-slab (A: 64x32, B: 128x32) as async 16B DMA chunks.
  auto issue = [&](int kk, int buf) {
    {
      const int row = tid >> 2, cb = (tid & 3) << 4;             // 64 B per A row
      unsigned lds = aBase[buf] + (unsigned)(row * 64 + cb);
      unsigned long long g =
          (unsigned long long)(uintptr_t)((const char*)A + ((size_t)(m0 + row) * K + kk) * 2 + cb);
      asm volatile("global_load_async_to_lds_b128 %0, %1, off" :: "v"(lds), "v"(g) : "memory");
    }
#pragma unroll
    for (int r = 0; r < 2; ++r) {
      const int idx = tid + (r << 8);                            // 512 chunks for B
      const int row = idx >> 2, cb = (idx & 3) << 4;
      unsigned lds = bBase[buf] + (unsigned)(row * 64 + cb);
      unsigned long long g =
          (unsigned long long)(uintptr_t)((const char*)WT + ((size_t)(n0 + row) * K + kk) * 2 + cb);
      asm volatile("global_load_async_to_lds_b128 %0, %1, off" :: "v"(lds), "v"(g) : "memory");
    }
  };

  v8f c[2][2];
  { v8f z = {}; c[0][0] = z; c[0][1] = z; c[1][0] = z; c[1][1] = z; }

  issue(0, 0);
  int s = 0;
  for (int kk = 0; kk < K; kk += 32, s ^= 1) {
    if (kk + 32 < K) {
      issue(kk + 32, s ^ 1);
      asm volatile("s_wait_asynccnt 0x3" ::: "memory");   // current slab landed
    } else {
      asm volatile("s_wait_asynccnt 0x0" ::: "memory");
    }
    __syncthreads();                                      // slab visible to all waves

    const bf16* Ab = As[s];
    const bf16* Bb = Bs[s];
    union { v16bf v; unsigned u[8]; } a[2], b[2];
#pragma unroll
    for (int i = 0; i < 8; ++i) {
      const int ka = aK(i, half), kb2 = aK(i, half);
      a[0].u[i] = *(const unsigned*)(Ab + ((wm << 5) +      r16) * 32 + ka);
      a[1].u[i] = *(const unsigned*)(Ab + ((wm << 5) + 16 + r16) * 32 + ka);
      b[0].u[i] = *(const unsigned*)(Bb + ((wn << 5) +      r16) * 32 + kb2);
      b[1].u[i] = *(const unsigned*)(Bb + ((wn << 5) + 16 + r16) * 32 + kb2);
    }
#pragma unroll
    for (int i = 0; i < 2; ++i)
#pragma unroll
      for (int j = 0; j < 2; ++j)
        c[i][j] = __builtin_amdgcn_wmma_f32_16x16x32_bf16(false, a[i].v, false, b[j].v,
                                                          (short)0, c[i][j], false, false);
    __syncthreads();                                      // safe to overwrite this buffer
  }

#pragma unroll
  for (int i = 0; i < 2; ++i)
#pragma unroll
    for (int j = 0; j < 2; ++j) {
      const int colg = n0 + (wn << 5) + (j << 4) + r16;
#pragma unroll
      for (int jj = 0; jj < 8; ++jj) {
        const int rowg = m0 + (wm << 5) + (i << 4) + jj + (half << 3);
        float v = c[i][j][jj];
        if (EPI >= 1) v += bias[colg];
        if (EPI == 1) v = 0.5f * v * (1.0f + erff(v * 0.70710678118654752f));  // exact GELU
        if (EPI == 2) {
          ((float*)outp)[(size_t)rowg * N + colg] = v + resid[(size_t)rowg * N + colg];
        } else {
          ((bf16*)outp)[(size_t)rowg * N + colg] = f2bf(v);
        }
      }
    }
}

// ---------------------------------------------------------------------------
// QKV prep: per-head LayerNorm of q and k over hd=64 (+SCALE on q), rearrange
// [tok][3][H][hd] -> q/k/v at [(batch*H + h)][n][hd].  One wave per (tok, h).
// ---------------------------------------------------------------------------
__global__ void qkv_prep(const bf16* __restrict__ QKV, const float* __restrict__ gw,
                         const float* __restrict__ gb, bf16* __restrict__ Qo,
                         bf16* __restrict__ Ko, bf16* __restrict__ Vo,
                         int nseq, int ntok) {
  const int lane = threadIdx.x & 31;
  const int gwv  = blockIdx.x * (blockDim.x >> 5) + (threadIdx.x >> 5);
  const int tok  = gwv >> 4;
  const int h    = gwv & 15;
  if (tok >= ntok) return;
  const bf16* base = QKV + (size_t)tok * (3 * C) + h * HD;
  __builtin_prefetch(base + (size_t)8 * 3 * C, 0, 1);   // global_prefetch_b8, 8 tokens ahead
  const int batch = tok / nseq, n = tok % nseq;
  const size_t dst = ((size_t)(batch * H + h) * nseq + n) * HD;

  Vo[dst + lane]      = base[2 * C + lane];
  Vo[dst + lane + 32] = base[2 * C + lane + 32];

#pragma unroll
  for (int which = 0; which < 2; ++which) {
    float x0 = bf2f(base[which * C + lane]);
    float x1 = bf2f(base[which * C + lane + 32]);
    float s = x0 + x1, sq = x0 * x0 + x1 * x1;
#pragma unroll
    for (int off = 16; off; off >>= 1) { s += __shfl_xor(s, off, 32); sq += __shfl_xor(sq, off, 32); }
    const float mean = s * (1.f / HD);
    const float var  = sq * (1.f / HD) - mean * mean;
    const float rs   = rsqrtf(var + 1e-5f);
    const float sc   = (which == 0) ? 0.125f : 1.f;   // SCALE = 8/64 on q
    bf16* O = (which == 0) ? Qo : Ko;
    O[dst + lane]      = f2bf(((x0 - mean) * rs * gw[lane]      + gb[lane])      * sc);
    O[dst + lane + 32] = f2bf(((x1 - mean) * rs * gw[lane + 32] + gb[lane + 32]) * sc);
  }
}

// ---------------------------------------------------------------------------
// Flash-style attention. One block per (batch, head). K/V staged into LDS via
// GLOBAL_LOAD_ASYNC_TO_LDS_B128 (ASYNCcnt), pad rows zero-filled and masked
// to -inf. Each wave owns 16 query rows; online softmax; scores and P@V both
// via WMMA bf16. P transposed through per-wave LDS scratch (same-wave DS ops
// are in-order on CDNA5, so no barrier needed in the chunk loop).
// Output token-major: O[(batch*nq + row)*C + h*64 + d].
// ---------------------------------------------------------------------------
__global__ __launch_bounds__(256, 1)
void attn_fwd(const bf16* __restrict__ Q, const bf16* __restrict__ Kp,
              const bf16* __restrict__ Vp, bf16* __restrict__ O,
              int nq, int nkv) {
  extern __shared__ unsigned smem_u[];
  const int lane = threadIdx.x & 31;
  const int wave = threadIdx.x >> 5;
  const int nwaves = blockDim.x >> 5;
  const int half = lane >> 4, r16 = lane & 15;
  const int bh = blockIdx.x;
  const int batch = bh / H, h = bh % H;
  const int nkp = (nkv + 31) & ~31;

  bf16* k_s = (bf16*)smem_u;
  bf16* v_s = k_s + (size_t)nkp * HD;
  bf16* p_s = v_s + (size_t)nkp * HD + (size_t)wave * 512;   // 16x32 per wave

  // ---- Async K/V stage: global -> LDS without VGPR round trip -------------
  {
    const char* Kg = (const char*)(Kp + (size_t)bh * nkv * HD);
    const char* Vg = (const char*)(Vp + (size_t)bh * nkv * HD);
    const unsigned kOff = (unsigned)(uintptr_t)k_s;   // LDS byte offset (low 32 bits)
    const unsigned vOff = (unsigned)(uintptr_t)v_s;
    const int validB = nkv * HD * 2;                  // bytes per matrix
    for (int off = (int)threadIdx.x * 16; off < validB; off += (int)blockDim.x * 16) {
      unsigned ldk = kOff + (unsigned)off;
      unsigned ldv = vOff + (unsigned)off;
      unsigned long long ga = (unsigned long long)(uintptr_t)(Kg + off);
      unsigned long long gv = (unsigned long long)(uintptr_t)(Vg + off);
      asm volatile("global_load_async_to_lds_b128 %0, %1, off" :: "v"(ldk), "v"(ga) : "memory");
      asm volatile("global_load_async_to_lds_b128 %0, %1, off" :: "v"(ldv), "v"(gv) : "memory");
    }
    // Zero-fill pad rows (temporal case: nkv=16 -> nkp=32).
    unsigned* ks = (unsigned*)k_s;
    unsigned* vs = (unsigned*)v_s;
    const int nwords = nkp * (HD / 2), vwords = nkv * (HD / 2);
    for (int w = vwords + (int)threadIdx.x; w < nwords; w += (int)blockDim.x) {
      ks[w] = 0u; vs[w] = 0u;
    }
    asm volatile("s_wait_asynccnt 0x0" ::: "memory");
  }
  __syncthreads();

  const int nrb = nq >> 4;
  for (int rb = wave; rb < nrb; rb += nwaves) {
    const int row0 = rb << 4;
    const unsigned* Qw = (const unsigned*)(Q + (size_t)(bh * nq + row0 + r16) * HD);
    union { v16bf v; unsigned u[8]; } q0, q1;
#pragma unroll
    for (int i = 0; i < 8; ++i) {
      const int ke = aK(i, half);
      q0.u[i] = Qw[ke >> 1];
      q1.u[i] = Qw[(32 + ke) >> 1];
    }
    float m[8], l[8];
    v8f acc[4];
#pragma unroll
    for (int j = 0; j < 8; ++j) { m[j] = -3.0e38f; l[j] = 0.f; }
    { v8f z = {};
#pragma unroll
      for (int t = 0; t < 4; ++t) acc[t] = z; }

    for (int kb = 0; kb < nkp; kb += 32) {
      v8f sc2[2];
#pragma unroll
      for (int sub = 0; sub < 2; ++sub) {
        const int n0 = kb + (sub << 4);
        const bf16* kr = k_s + (size_t)(n0 + r16) * HD;
        union { v16bf v; unsigned u[8]; } b0, b1;
#pragma unroll
        for (int i = 0; i < 8; ++i) {
          const int ke = aK(i, half);
          b0.u[i] = *(const unsigned*)(kr + ke);
          b1.u[i] = *(const unsigned*)(kr + 32 + ke);
        }
        v8f sv = {};
        sv = __builtin_amdgcn_wmma_f32_16x16x32_bf16(false, q0.v, false, b0.v, (short)0, sv, false, false);
        sv = __builtin_amdgcn_wmma_f32_16x16x32_bf16(false, q1.v, false, b1.v, (short)0, sv, false, false);
        if (n0 + r16 >= nkv) {
#pragma unroll
          for (int j = 0; j < 8; ++j) sv[j] = -3.0e38f;   // mask pad keys
        }
        sc2[sub] = sv;
      }
      float scl[8];
#pragma unroll
      for (int j = 0; j < 8; ++j) {
        float mx = fmaxf(sc2[0][j], sc2[1][j]);
#pragma unroll
        for (int off = 8; off; off >>= 1) mx = fmaxf(mx, __shfl_xor(mx, off, 32));
        const float mn = fmaxf(m[j], mx);
        const float p0 = expf(sc2[0][j] - mn);
        const float p1 = expf(sc2[1][j] - mn);
        float rs = p0 + p1;
#pragma unroll
        for (int off = 8; off; off >>= 1) rs += __shfl_xor(rs, off, 32);
        scl[j] = expf(m[j] - mn);
        l[j] = l[j] * scl[j] + rs;
        m[j] = mn;
        const int prow = j + (half << 3);
        p_s[prow * 32 + r16]      = f2bf(p0);
        p_s[prow * 32 + 16 + r16] = f2bf(p1);
      }
#pragma unroll
      for (int t = 0; t < 4; ++t)
#pragma unroll
        for (int j = 0; j < 8; ++j) acc[t][j] *= scl[j];

      // Reload P in A-fragment layout (same-wave LDS RAW: in-order).
      union { v16bf v; unsigned u[8]; } pf;
#pragma unroll
      for (int i = 0; i < 8; ++i)
        pf.u[i] = *(const unsigned*)(p_s + r16 * 32 + aK(i, half));

#pragma unroll
      for (int t = 0; t < 4; ++t) {
        const int d = (t << 4) + r16;
        union { v16bf v; unsigned u[8]; } vf;
#pragma unroll
        for (int i = 0; i < 8; ++i) {
          const int k0 = kb + aK(i, half);
          const unsigned lo = bf_bits(v_s[(size_t)k0 * HD + d]);
          const unsigned hi = bf_bits(v_s[(size_t)(k0 + 1) * HD + d]);
          vf.u[i] = lo | (hi << 16);
        }
        acc[t] = __builtin_amdgcn_wmma_f32_16x16x32_bf16(false, pf.v, false, vf.v,
                                                         (short)0, acc[t], false, false);
      }
    }
    // Normalize and store token-major.
#pragma unroll
    for (int t = 0; t < 4; ++t)
#pragma unroll
      for (int j = 0; j < 8; ++j) {
        const int row = row0 + j + (half << 3);
        O[((size_t)(batch * nq + row)) * C + h * HD + (t << 4) + r16] = f2bf(acc[t][j] / l[j]);
      }
  }
}

// ---------------------------------------------------------------------------
// [B,D1,D2,C] -> [B,D2,D1,C] row-wise transpose copy (f32, float4 moves).
// ---------------------------------------------------------------------------
__global__ void seq_transpose(const float* __restrict__ in, float* __restrict__ out,
                              int D1, int D2) {
  const int rowid = blockIdx.x;
  const int b = rowid / (D1 * D2);
  const int r = rowid % (D1 * D2);
  const int i = r / D2, j = r % D2;
  const float4* src = (const float4*)(in + (size_t)rowid * C);
  float4* dst = (float4*)(out + ((size_t)((b * D2 + j) * D1 + i)) * C);
  for (int cidx = threadIdx.x; cidx < C / 4; cidx += blockDim.x) dst[cidx] = src[cidx];
}

// ---------------------------------------------------------------------------
extern "C" void kernel_launch(void* const* d_in, const int* in_sizes, int n_in,
                              void* d_out, int out_size, void* d_ws, size_t ws_size,
                              hipStream_t stream) {
  const float* x        = (const float*)d_in[0];
  const float* ns_w     = (const float*)d_in[1];
  const float* ns_b     = (const float*)d_in[2];
  const float* nt_w     = (const float*)d_in[3];
  const float* nt_b     = (const float*)d_in[4];
  const float* nm_w     = (const float*)d_in[5];
  const float* nm_b     = (const float*)d_in[6];
  const float* s_qkv    = (const float*)d_in[7];
  const float* s_qkn_w  = (const float*)d_in[8];
  const float* s_qkn_b  = (const float*)d_in[9];
  const float* s_proj_w = (const float*)d_in[10];
  const float* s_proj_b = (const float*)d_in[11];
  const float* t_qkv    = (const float*)d_in[12];
  const float* t_qkn_w  = (const float*)d_in[13];
  const float* t_qkn_b  = (const float*)d_in[14];
  const float* t_proj_w = (const float*)d_in[15];
  const float* t_proj_b = (const float*)d_in[16];
  const float* fc1_w    = (const float*)d_in[17];
  const float* fc1_b    = (const float*)d_in[18];
  const float* fc2_w    = (const float*)d_in[19];
  const float* fc2_b    = (const float*)d_in[20];
  float* out = (float*)d_out;

  // Workspace carve-up (~252 MB).
  char* p = (char*)d_ws;
  auto alloc = [&](size_t bytes) -> char* {
    char* r = p; p += (bytes + 255) & ~(size_t)255; return r;
  };
  bf16* wt_s_qkv  = (bf16*)alloc((size_t)3 * C * C * 2);
  bf16* wt_s_proj = (bf16*)alloc((size_t)C * C * 2);
  bf16* wt_t_qkv  = (bf16*)alloc((size_t)3 * C * C * 2);
  bf16* wt_t_proj = (bf16*)alloc((size_t)C * C * 2);
  bf16* wt_fc1    = (bf16*)alloc((size_t)C * HID * 2);
  bf16* wt_fc2    = (bf16*)alloc((size_t)C * HID * 2);
  bf16* lnbuf     = (bf16*)alloc((size_t)NTOK * C * 2);
  bf16* bigbuf    = (bf16*)alloc((size_t)NTOK * HID * 2);  // qkv (48MB) then h (64MB)
  bf16* qb        = (bf16*)alloc((size_t)NTOK * C * 2);
  bf16* kb        = (bf16*)alloc((size_t)NTOK * C * 2);
  bf16* vb        = (bf16*)alloc((size_t)NTOK * C * 2);
  bf16* attnbuf   = (bf16*)alloc((size_t)NTOK * C * 2);
  float* f32a     = (float*)alloc((size_t)NTOK * C * 4);
  float* f32b     = (float*)alloc((size_t)NTOK * C * 4);

  // --- 0. weights to bf16, transposed to [N][K] ---
  wt_cvt<<<3 * C, 256, 0, stream>>>(s_qkv,    wt_s_qkv,  C,   3 * C);
  wt_cvt<<<C,     256, 0, stream>>>(s_proj_w, wt_s_proj, C,   C);
  wt_cvt<<<3 * C, 256, 0, stream>>>(t_qkv,    wt_t_qkv,  C,   3 * C);
  wt_cvt<<<C,     256, 0, stream>>>(t_proj_w, wt_t_proj, C,   C);
  wt_cvt<<<HID,   256, 0, stream>>>(fc1_w,    wt_fc1,    C,   HID);
  wt_cvt<<<C,     256, 0, stream>>>(fc2_w,    wt_fc2,    HID, C);

  const dim3 gQKV(3 * C / 128, NTOK / 64), gC(C / 128, NTOK / 64), gH(HID / 128, NTOK / 64);
  const size_t lds_sp = ((size_t)256 * HD * 2 * 2) + 8 * 512 * 2;   // 73728 B
  const size_t lds_tp = ((size_t)32  * HD * 2 * 2) + 1 * 512 * 2;   //  9216 B

  // --- 1. spatial attention (sequences of S=256, batch B*T=32) ---
  ln_rows<<<NTOK / 8, 256, 0, stream>>>(x, ns_w, ns_b, lnbuf, NTOK);
  gemm_bf16<0><<<gQKV, 256, 0, stream>>>(lnbuf, wt_s_qkv, nullptr, nullptr, bigbuf, NTOK, 3 * C, C);
  qkv_prep<<<NTOK * H / 8, 256, 0, stream>>>(bigbuf, s_qkn_w, s_qkn_b, qb, kb, vb, 256, NTOK);
  attn_fwd<<<32 * H, 256, lds_sp, stream>>>(qb, kb, vb, attnbuf, 256, 256);
  gemm_bf16<2><<<gC, 256, 0, stream>>>(attnbuf, wt_s_proj, s_proj_b, x, f32a, NTOK, C, C);

  // --- 2. temporal attention (sequences of T=16, batch B*S=512) ---
  seq_transpose<<<NTOK, 256, 0, stream>>>(f32a, f32b, 16, 256);           // [B,T,S]->[B,S,T]
  ln_rows<<<NTOK / 8, 256, 0, stream>>>(f32b, nt_w, nt_b, lnbuf, NTOK);
  gemm_bf16<0><<<gQKV, 256, 0, stream>>>(lnbuf, wt_t_qkv, nullptr, nullptr, bigbuf, NTOK, 3 * C, C);
  qkv_prep<<<NTOK * H / 8, 256, 0, stream>>>(bigbuf, t_qkn_w, t_qkn_b, qb, kb, vb, 16, NTOK);
  attn_fwd<<<512 * H, 32, lds_tp, stream>>>(qb, kb, vb, attnbuf, 16, 16);
  gemm_bf16<2><<<gC, 256, 0, stream>>>(attnbuf, wt_t_proj, t_proj_b, f32b, f32a, NTOK, C, C);
  seq_transpose<<<NTOK, 256, 0, stream>>>(f32a, f32b, 256, 16);           // [B,S,T]->[B,T,S]

  // --- 3. MLP: x + fc2(gelu(fc1(LN(x)))) ---
  ln_rows<<<NTOK / 8, 256, 0, stream>>>(f32b, nm_w, nm_b, lnbuf, NTOK);
  gemm_bf16<1><<<gH, 256, 0, stream>>>(lnbuf, wt_fc1, fc1_b, nullptr, bigbuf, NTOK, HID, C);
  gemm_bf16<2><<<gC, 256, 0, stream>>>(bigbuf, wt_fc2, fc2_b, f32b, out, NTOK, C, HID);

  (void)in_sizes; (void)n_in; (void)out_size; (void)ws_size;
}
